// Multi_Head_Attention_59596966199577
// MI455X (gfx1250) — matrix operand order repaired and checked by reading, not compiled, not run on hardware
//
#include <hip/hip_runtime.h>
#include <hip/hip_bf16.h>

typedef __attribute__((ext_vector_type(16))) _Float16 v16h;
typedef __attribute__((ext_vector_type(8)))  _Float16 v8h;
typedef __attribute__((ext_vector_type(8)))  float    v8f;
typedef __attribute__((ext_vector_type(4)))  unsigned int v4u;
typedef __attribute__((ext_vector_type(4)))  int v4i;
typedef __attribute__((ext_vector_type(8)))  int v8i;

#define S_LEN  2048
#define DMODEL 512
#define DKV    64
#define NHEAD  8
#define BATCH  4

// ---------------------------------------------------------------------------
// WMMA wrapper: D = A(16x32 f16) * B(32x16 f16) + C(16x16 f32)
// ---------------------------------------------------------------------------
__device__ __forceinline__ v8f wmma_f16(v16h a, v16h b, v8f c) {
  return __builtin_amdgcn_wmma_f32_16x16x32_f16(
      false, a, false, b, (short)0, c, false, false);
}

__device__ __forceinline__ void wave_lds_fence() {
  asm volatile("s_wait_dscnt 0x0" ::: "memory");
}

// LDS byte offset of a __shared__ object: generic LDS pointers carry the
// wave-relative LDS byte address in their low 32 bits (flat aperture rule).
__device__ __forceinline__ unsigned int lds_off(const void* p) {
  return (unsigned int)(unsigned long long)p;
}

// ---------------------------------------------------------------------------
// Tensor Data Mover: 2-D tile load global->LDS (f16 elements, data_size=2B).
// D# per CDNA5 ISA ch.8:  group0 = {count/type/lds_addr/global_addr},
// group1 = {data_size, tensor_dim0/1, tile_dim0/1, dim0_stride}.
// ---------------------------------------------------------------------------
__device__ __forceinline__ void tdm_load_2d_f16(
    const void* gptr, unsigned int lds_byte,
    unsigned int tile_w,   // elements per tile row
    unsigned int tile_h,   // tile rows
    unsigned int stride,   // tensor row stride (elements)
    unsigned int tens_w,   // tensor row length (elements)
    unsigned int tens_h) { // tensor rows
  unsigned long long ga = (unsigned long long)gptr;
  v4u g0;
  g0.x = 1u;                                   // count=1, user descriptor
  g0.y = lds_byte;                             // LDS dest byte address
  g0.z = (unsigned int)ga;                     // global_addr[31:0]
  g0.w = ((unsigned int)(ga >> 32) & 0x01FFFFFFu) | (2u << 30);  // addr[56:32] | type=2

  unsigned int w0 = (1u << 16);                // workgroup_mask=0, data_size=1 (2B)
  unsigned int w1 = (tens_w & 0xFFFFu) << 16;  // tensor_dim0[15:0] @ bits 63:48
  unsigned int w2 = ((tens_w >> 16) & 0xFFFFu) | ((tens_h & 0xFFFFu) << 16);
  unsigned int w3 = ((tens_h >> 16) & 0xFFFFu) | ((tile_w & 0xFFFFu) << 16);
  unsigned int w4 = (tile_h & 0xFFFFu);        // tile_dim1; tile_dim2 = 0
  unsigned int w5 = stride;                    // tensor_dim0_stride[31:0]
  unsigned int w6 = 0;                         // stride[47:32]=0, dim1_stride lo = 0
  unsigned int w7 = 0;
  v8i g1 = {(int)w0, (int)w1, (int)w2, (int)w3,
            (int)w4, (int)w5, (int)w6, (int)w7};
  v4i z4 = {0, 0, 0, 0};
#if defined(__clang_major__) && (__clang_major__ >= 23)
  v8i z8 = {0, 0, 0, 0, 0, 0, 0, 0};
  __builtin_amdgcn_tensor_load_to_lds(g0, g1, z4, z4, z8, 0);
#else
  __builtin_amdgcn_tensor_load_to_lds(g0, g1, z4, z4, 0);
#endif
}

// ---------------------------------------------------------------------------
// Fragment loaders (CDNA5 ISA 7.12.2 wave32 layouts).
// ---------------------------------------------------------------------------

// A fragment (16x32 f16) from row-major fp32 with on-the-fly convert
__device__ __forceinline__ v16h frag_a_f32(const float* __restrict__ base,
                                           int ld, int m0, int k0) {
  const int lane = threadIdx.x & 31;
  const int row  = m0 + (lane & 15);
  const int khi  = (lane < 16) ? 0 : 8;
  const float* p = base + (size_t)row * ld + k0;
  v8f lo = *(const v8f*)(p + khi);
  v8f hi = *(const v8f*)(p + 16 + khi);
  v8h lo16 = __builtin_convertvector(lo, v8h);
  v8h hi16 = __builtin_convertvector(hi, v8h);
  return __builtin_shufflevector(lo16, hi16,
      0, 1, 2, 3, 4, 5, 6, 7, 8, 9, 10, 11, 12, 13, 14, 15);
}

// A fragment from row-major f16 (global or LDS pointer)
__device__ __forceinline__ v16h frag_a_f16(const _Float16* __restrict__ base,
                                           int ld, int m0, int k0) {
  const int lane = threadIdx.x & 31;
  const int row  = m0 + (lane & 15);
  const int khi  = (lane < 16) ? 0 : 8;
  const _Float16* p = base + (size_t)row * ld + k0;
  v8h lo = *(const v8h*)(p + khi);
  v8h hi = *(const v8h*)(p + 16 + khi);
  return __builtin_shufflevector(lo, hi,
      0, 1, 2, 3, 4, 5, 6, 7, 8, 9, 10, 11, 12, 13, 14, 15);
}

// B fragment of M^T where M is row-major f16 [N x ld]: B(k,n) = M[n][k].
// Two 8-half loads per lane (k = 8h+s, 16+8h+s). Works for global or LDS pointers.
__device__ __forceinline__ v16h frag_bT_f16(const _Float16* __restrict__ base,
                                            int ld, int n0, int k0) {
  const int lane = threadIdx.x & 31;
  const int nrow = n0 + (lane & 15);
  const int kofs = (lane < 16) ? 0 : 8;
  const _Float16* p = base + (size_t)nrow * ld + k0 + kofs;
  v8h lo = *(const v8h*)(p);
  v8h hi = *(const v8h*)(p + 16);
  return __builtin_shufflevector(lo, hi,
      0, 1, 2, 3, 4, 5, 6, 7, 8, 9, 10, 11, 12, 13, 14, 15);
}

// ---------------------------------------------------------------------------
// Kernel 0a: convert per-head projection weights to f16, transposed:
//   wT[h][n][k] = W[h][k][n]   ([H, 64, 512])   grid = (H, 3), block = 256
// ---------------------------------------------------------------------------
__global__ __launch_bounds__(256) void convert_w_kernel(
    const float* __restrict__ Wq, const float* __restrict__ Wk,
    const float* __restrict__ Wv, _Float16* __restrict__ wqT,
    _Float16* __restrict__ wkT, _Float16* __restrict__ wvT) {
  const float* src;
  _Float16*    dst;
  if (blockIdx.y == 0)      { src = Wq; dst = wqT; }
  else if (blockIdx.y == 1) { src = Wk; dst = wkT; }
  else                      { src = Wv; dst = wvT; }
  const int h = blockIdx.x;
  src += (size_t)h * DMODEL * DKV;
  dst += (size_t)h * DKV * DMODEL;
  for (int idx = threadIdx.x; idx < DKV * DMODEL; idx += 256) {
    const int n = idx >> 9;      // / DMODEL
    const int k = idx & 511;     // % DMODEL
    dst[idx] = (_Float16)src[(size_t)k * DKV + n];
  }
}

// Kernel 0b: woT[n][k] = Wo[k][n]   ([512, 512])   grid = 64, block = 256
__global__ __launch_bounds__(256) void convert_wo_kernel(
    const float* __restrict__ Wo, _Float16* __restrict__ woT) {
  const int total = DMODEL * DMODEL;
  for (int idx = blockIdx.x * 256 + threadIdx.x; idx < total;
       idx += gridDim.x * 256) {
    const int n = idx >> 9;
    const int k = idx & 511;
    woT[idx] = (_Float16)Wo[(size_t)k * DMODEL + n];
  }
}

// ---------------------------------------------------------------------------
// Kernel 1: Q/K/V projections. grid = (S/64, B*H, 3), block = 128 (4 waves).
// B fragments come from pre-transposed f16 weights (contiguous loads).
// V is written TRANSPOSED: vT[b,h][vcol][s]  -> contiguous B-frags later.
// ---------------------------------------------------------------------------
__global__ __launch_bounds__(128) void qkv_proj_kernel(
    const float* __restrict__ x,
    const _Float16* __restrict__ wqT, const _Float16* __restrict__ wkT,
    const _Float16* __restrict__ wvT,
    _Float16* __restrict__ qh, _Float16* __restrict__ kh,
    _Float16* __restrict__ vt) {
  const int wave = threadIdx.x >> 5;
  const int lane = threadIdx.x & 31;
  const int bh = blockIdx.y;
  const int b  = bh >> 3;
  const int h  = bh & 7;
  const int m0 = blockIdx.x * 64 + wave * 16;
  (void)b;

  const _Float16* WT = (blockIdx.z == 0) ? wqT : (blockIdx.z == 1) ? wkT : wvT;
  const float*    xb  = x + (size_t)(bh >> 3) * S_LEN * DMODEL;
  const _Float16* WTh = WT + (size_t)h * DKV * DMODEL;

  v8f acc[4] = {};
  for (int k0 = 0; k0 < DMODEL; k0 += 32) {
    v16h a = frag_a_f32(xb, DMODEL, m0, k0);
#pragma unroll
    for (int t = 0; t < 4; ++t)
      acc[t] = wmma_f16(a, frag_bT_f16(WTh, DMODEL, t * 16, k0), acc[t]);
  }

  const int col  = lane & 15;
  const int mofs = (lane < 16) ? 0 : 8;
  if (blockIdx.z == 2) {
    _Float16* vb = vt + (size_t)bh * DKV * S_LEN;   // [64][S]
#pragma unroll
    for (int t = 0; t < 4; ++t)
#pragma unroll
      for (int i = 0; i < 8; ++i)
        vb[(size_t)(t * 16 + col) * S_LEN + (m0 + mofs + i)] =
            (_Float16)acc[t][i];
  } else {
    _Float16* ob = ((blockIdx.z == 0) ? qh : kh) + (size_t)bh * S_LEN * DKV;
#pragma unroll
    for (int t = 0; t < 4; ++t)
#pragma unroll
      for (int i = 0; i < 8; ++i)
        ob[(size_t)(m0 + mofs + i) * DKV + t * 16 + col] = (_Float16)acc[t][i];
  }
}

// ---------------------------------------------------------------------------
// Kernel 2: flash attention per (b,h). grid = (S/64, B*H), block = 128.
// K and V^T tiles (64x64 f16) are staged into LDS by the Tensor Data Mover,
// double-buffered: wave 0 issues tensor_load_to_lds for tile i+1, waits
// s_wait_tensorcnt<=2 for tile i, then all 4 waves compute from LDS.
// ---------------------------------------------------------------------------
__global__ __launch_bounds__(128) void attn_kernel(
    const _Float16* __restrict__ qh, const _Float16* __restrict__ kh,
    const _Float16* __restrict__ vtg, _Float16* __restrict__ oh) {
  __shared__ __align__(16) _Float16 s_k[2][64][64];    // K tile  [key][c]
  __shared__ __align__(16) _Float16 s_vt[2][64][64];   // V^T tile [vcol][key]
  __shared__ float    s_scores[4][16][64];
  __shared__ _Float16 s_p[4][16][64];
  __shared__ float    s_scale[4][16];

  const int wave = threadIdx.x >> 5;
  const int lane = threadIdx.x & 31;
  const int bh = blockIdx.y;
  const int b  = bh >> 3;
  const int h  = bh & 7;
  const int q0 = blockIdx.x * 64 + wave * 16;

  const _Float16* Qb  = qh  + (size_t)bh * S_LEN * DKV;
  const _Float16* Kb  = kh  + (size_t)bh * S_LEN * DKV;
  const _Float16* Vtb = vtg + (size_t)bh * DKV * S_LEN;   // [64][S]

  // Q fragments loaded once (d_k = 64 -> two k-steps of 32)
  const v16h aq0 = frag_a_f16(Qb, DKV, q0, 0);
  const v16h aq1 = frag_a_f16(Qb, DKV, q0, 32);

  // Preload tile 0 into buffer 0 via TDM (wave 0 only).
  if (wave == 0) {
    tdm_load_2d_f16(Kb,  lds_off(&s_k[0][0][0]),  64, 64, DKV,   DKV,   S_LEN);
    tdm_load_2d_f16(Vtb, lds_off(&s_vt[0][0][0]), 64, 64, S_LEN, S_LEN, DKV);
  }

  v8f oacc[4] = {};
  float m_i = -1e30f, l_i = 0.0f;
  const int col  = lane & 15;
  const int mofs = (lane < 16) ? 0 : 8;
  const float scale = 0.125f;   // 1/sqrt(64)

  for (int it = 0; it < S_LEN / 64; ++it) {
    const int kt  = it * 64;
    const int buf = it & 1;
    if (wave == 0) {
      if (kt + 64 < S_LEN) {
        const int nb = buf ^ 1;
        tdm_load_2d_f16(Kb + (size_t)(kt + 64) * DKV,
                        lds_off(&s_k[nb][0][0]), 64, 64, DKV, DKV, S_LEN);
        tdm_load_2d_f16(Vtb + (kt + 64),
                        lds_off(&s_vt[nb][0][0]), 64, 64, S_LEN, S_LEN, DKV);
        __builtin_amdgcn_s_wait_tensorcnt(2);   // tile `it` complete
      } else {
        __builtin_amdgcn_s_wait_tensorcnt(0);
      }
    }
    __syncthreads();

    // ---- scores tile S = (Q K^T) * scale, 16x64, into LDS ----
    const _Float16* kb = &s_k[buf][0][0];
#pragma unroll
    for (int t = 0; t < 4; ++t) {
      v8f s = {};
      s = wmma_f16(aq0, frag_bT_f16(kb, 64, t * 16, 0),  s);
      s = wmma_f16(aq1, frag_bT_f16(kb, 64, t * 16, 32), s);
#pragma unroll
      for (int i = 0; i < 8; ++i)
        s_scores[wave][mofs + i][t * 16 + col] = s[i] * scale;
    }
    wave_lds_fence();

    // ---- online softmax: lanes 0..15 each own one query row ----
    if (lane < 16) {
      float mt = m_i;
      for (int c = 0; c < 64; ++c)
        mt = fmaxf(mt, s_scores[wave][lane][c]);
      const float fac = __expf(m_i - mt);
      float sum = 0.0f;
      for (int c = 0; c < 64; ++c) {
        const float p = __expf(s_scores[wave][lane][c] - mt);
        s_p[wave][lane][c] = (_Float16)p;
        sum += p;
      }
      l_i = l_i * fac + sum;
      m_i = mt;
      s_scale[wave][lane] = fac;
    }
    wave_lds_fence();

    // ---- rescale running accumulator by per-row factor ----
    float f[8];
#pragma unroll
    for (int i = 0; i < 8; ++i) f[i] = s_scale[wave][mofs + i];
#pragma unroll
    for (int t = 0; t < 4; ++t)
#pragma unroll
      for (int i = 0; i < 8; ++i) oacc[t][i] *= f[i];

    // ---- O += P * V  (P and V^T both from LDS, contiguous B-frags) ----
    const v16h ap0 = frag_a_f16(&s_p[wave][0][0], 64, 0, 0);
    const v16h ap1 = frag_a_f16(&s_p[wave][0][0], 64, 0, 32);
    const _Float16* vb = &s_vt[buf][0][0];
#pragma unroll
    for (int t = 0; t < 4; ++t) {
      oacc[t] = wmma_f16(ap0, frag_bT_f16(vb, 64, t * 16, 0),  oacc[t]);
      oacc[t] = wmma_f16(ap1, frag_bT_f16(vb, 64, t * 16, 32), oacc[t]);
    }
    __syncthreads();   // all waves done with buffer `buf` before TDM reuses it
  }

  // ---- finalize: O /= l, store f16 in head-concat layout [B,S,H*64] ----
  if (lane < 16) s_scale[wave][lane] = 1.0f / l_i;
  wave_lds_fence();
  float f[8];
#pragma unroll
  for (int i = 0; i < 8; ++i) f[i] = s_scale[wave][mofs + i];
#pragma unroll
  for (int t = 0; t < 4; ++t)
#pragma unroll
    for (int i = 0; i < 8; ++i) {
      const size_t row = (size_t)b * S_LEN + q0 + mofs + i;
      oh[row * DMODEL + h * DKV + t * 16 + col] = (_Float16)(oacc[t][i] * f[i]);
    }
}

// ---------------------------------------------------------------------------
// Kernel 3: output projection [B*S,512] x Wo[512,512] -> fp32 out.
// grid = (B*S/64, 512/64), block = 128. B-frags from pre-transposed f16 Wo.
// ---------------------------------------------------------------------------
__global__ __launch_bounds__(128) void out_proj_kernel(
    const _Float16* __restrict__ oh, const _Float16* __restrict__ woT,
    float* __restrict__ out) {
  const int wave = threadIdx.x >> 5;
  const int lane = threadIdx.x & 31;
  const int m0  = blockIdx.x * 64 + wave * 16;
  const int n0b = blockIdx.y * 64;

  v8f acc[4] = {};
  for (int k0 = 0; k0 < DMODEL; k0 += 32) {
    v16h a = frag_a_f16(oh, DMODEL, m0, k0);
#pragma unroll
    for (int t = 0; t < 4; ++t)
      acc[t] = wmma_f16(a, frag_bT_f16(woT, DMODEL, n0b + t * 16, k0), acc[t]);
  }

  const int col  = lane & 15;
  const int mofs = (lane < 16) ? 0 : 8;
#pragma unroll
  for (int t = 0; t < 4; ++t)
#pragma unroll
    for (int i = 0; i < 8; ++i)
      out[(size_t)(m0 + mofs + i) * DMODEL + n0b + t * 16 + col] = acc[t][i];
}

// ---------------------------------------------------------------------------
extern "C" void kernel_launch(void* const* d_in, const int* in_sizes, int n_in,
                              void* d_out, int out_size, void* d_ws,
                              size_t ws_size, hipStream_t stream) {
  const float* x  = (const float*)d_in[0];
  const float* Wq = (const float*)d_in[1];
  const float* Wk = (const float*)d_in[2];
  const float* Wv = (const float*)d_in[3];
  const float* Wo = (const float*)d_in[4];
  float* out = (float*)d_out;

  // workspace (f16): Q,K [B,H,S,64]; V^T [B,H,64,S]; concat O [B,S,512];
  // transposed f16 weights.
  const size_t NQ = (size_t)BATCH * NHEAD * S_LEN * DKV;   // 4,194,304 halves
  const size_t NW = (size_t)NHEAD * DKV * DMODEL;          //   262,144 halves
  _Float16* qh  = (_Float16*)d_ws;
  _Float16* kh  = qh + NQ;
  _Float16* vt  = kh + NQ;
  _Float16* oh  = vt + NQ;
  _Float16* wqT = oh + NQ;
  _Float16* wkT = wqT + NW;
  _Float16* wvT = wkT + NW;
  _Float16* woT = wvT + NW;   // 512*512 halves

  convert_w_kernel<<<dim3(NHEAD, 3), 256, 0, stream>>>(Wq, Wk, Wv,
                                                       wqT, wkT, wvT);
  convert_wo_kernel<<<dim3(64), 256, 0, stream>>>(Wo, woT);
  qkv_proj_kernel<<<dim3(S_LEN / 64, BATCH * NHEAD, 3), 128, 0, stream>>>(
      x, wqT, wkT, wvT, qh, kh, vt);
  attn_kernel<<<dim3(S_LEN / 64, BATCH * NHEAD), 128, 0, stream>>>(
      qh, kh, vt, oh);
  out_proj_kernel<<<dim3((BATCH * S_LEN) / 64, DMODEL / 64), 128, 0, stream>>>(
      oh, woT, out);
}
